// GCN_22539988370028
// MI455X (gfx1250) — matrix operand order repaired and checked by reading, not compiled, not run on hardware
//
#include <hip/hip_runtime.h>
#include <cstdint>

typedef __attribute__((ext_vector_type(16))) _Float16 v16h;
typedef __attribute__((ext_vector_type(8)))  _Float16 v8h;
typedef __attribute__((ext_vector_type(4)))  _Float16 v4h;
typedef __attribute__((ext_vector_type(8)))  float    v8f;
typedef __attribute__((ext_vector_type(4)))  float    v4f;
typedef __attribute__((ext_vector_type(4)))  int      v4i;

#define GCN_N 512
#define GCN_B 64
#define TILE  128
#define LDSS  40   // LDS row stride in f16: 32 data + 8 pad -> 80B rows (16B aligned, conflict-free)

#define AS1 __attribute__((address_space(1)))
#define AS3 __attribute__((address_space(3)))

// Async global->LDS B128 copy (CDNA5 GLOBAL_LOAD_ASYNC_TO_LDS_B128, ASYNCcnt-tracked).
// Probe-discovered signature: (v4i global*, v4i lds*, imm offset, imm cpol).
#if defined(__gfx1250__) && __has_builtin(__builtin_amdgcn_global_load_async_to_lds_b128)
#define GCN_ASYNC 1
__device__ __forceinline__ void gcn_async_cp16(void* lds, const void* g) {
    __builtin_amdgcn_global_load_async_to_lds_b128(
        (AS1 v4i*)(uintptr_t)g,                 // global src
        (AS3 v4i*)(uint32_t)(uintptr_t)lds,     // LDS dst (low 32 bits = LDS offset)
        0, 0);
}
__device__ __forceinline__ void gcn_wait_async() {
#if __has_builtin(__builtin_amdgcn_s_wait_asynccnt)
    __builtin_amdgcn_s_wait_asynccnt(0);
#else
    asm volatile("s_wait_asynccnt 0x0" ::: "memory");
#endif
}
#else
#define GCN_ASYNC 0
#endif

// ---------------------------------------------------------------------------
// r[b,j] = rsqrt(sum_i adj[b,i,j]), 0 if sum == 0
// ---------------------------------------------------------------------------
__global__ __launch_bounds__(256)
void gcn_colsum_rsqrt(const float* __restrict__ adj, float* __restrict__ r) {
    const int N = GCN_N;
    const int b = blockIdx.x;
    const float* A = adj + (size_t)b * N * N;
    for (int j = threadIdx.x; j < N; j += 256) {
        float s = 0.f;
        for (int i = 0; i < N; ++i) s += A[(size_t)i * N + j];
        r[b * N + j] = (s > 0.f) ? rsqrtf(s) : 0.f;
    }
}

// ---------------------------------------------------------------------------
// Ahat[b,i,j] = r[i] * adj[b,j,i] * r[j]  (32x32 LDS tile transpose, f16 out)
// ---------------------------------------------------------------------------
__global__ __launch_bounds__(256)
void gcn_build_ahat(const float* __restrict__ adj, const float* __restrict__ r,
                    _Float16* __restrict__ Ah) {
    const int N = GCN_N;
    __shared__ float t[32][33];
    const int b  = blockIdx.y;
    const int nb = N / 32;
    const int i0 = (blockIdx.x % nb) * 32;
    const int j0 = (blockIdx.x / nb) * 32;
    const int tx = threadIdx.x & 31;
    const int ty = threadIdx.x >> 5;       // 8 rows of 32
    const float* A  = adj + (size_t)b * N * N;
    const float* rb = r + b * N;
    #pragma unroll
    for (int row = ty; row < 32; row += 8)
        t[row][tx] = A[(size_t)(j0 + row) * N + i0 + tx];   // coalesced in tx
    __syncthreads();
    #pragma unroll
    for (int row = ty; row < 32; row += 8) {
        float v = rb[i0 + row] * t[tx][row] * rb[j0 + tx];
        Ah[((size_t)b * N + i0 + row) * N + j0 + tx] = (_Float16)v;  // coalesced in tx
    }
}

// ---------------------------------------------------------------------------
// f32 -> f16 elementwise cast, 4-wide
// ---------------------------------------------------------------------------
__global__ __launch_bounds__(256)
void gcn_cast_h4(const float* __restrict__ s, _Float16* __restrict__ d, size_t n4) {
    size_t i = (size_t)blockIdx.x * 256 + threadIdx.x;
    if (i < n4) {
        v4f v = ((const v4f*)s)[i];
        v4h o;
        o[0] = (_Float16)v[0]; o[1] = (_Float16)v[1];
        o[2] = (_Float16)v[2]; o[3] = (_Float16)v[3];
        ((v4h*)d)[i] = o;
    }
}

// ---------------------------------------------------------------------------
// Wt[n][k] = (f16) W[k][n]   (32x32 LDS tile transpose)
// ---------------------------------------------------------------------------
__global__ __launch_bounds__(256)
void gcn_transpose_w(const float* __restrict__ W, _Float16* __restrict__ Wt) {
    const int N = GCN_N;
    __shared__ float t[32][33];
    const int nb = N / 32;
    const int n0 = (blockIdx.x % nb) * 32;
    const int k0 = (blockIdx.x / nb) * 32;
    const int tx = threadIdx.x & 31;
    const int ty = threadIdx.x >> 5;
    #pragma unroll
    for (int row = ty; row < 32; row += 8)
        t[row][tx] = W[(size_t)(k0 + row) * N + n0 + tx];
    __syncthreads();
    #pragma unroll
    for (int row = ty; row < 32; row += 8)
        Wt[(size_t)(n0 + row) * N + k0 + tx] = (_Float16)t[tx][row];
}

// ---------------------------------------------------------------------------
// WMMA GEMM:  Out = Aop (row-major [m][k]) x B (columns given as rows of Bop)
//   Bop row n holds column n of the logical B matrix (i.e. Bt[n][k]).
// Block: 128x128 tile, 256 threads = 8 waves (4 along M x 2 along N).
// Wave:  32x64 = 2x4 grid of 16x16 tiles, K-step 32 (v_wmma_f32_16x16x32_f16).
// Staging: double-buffered LDS; async global->LDS copies (ASYNCcnt) when the
// toolchain exposes them, else global_load_b128 + ds_store_b128 (+prefetch).
// MODE 0: store OutT (f16, out[n][m]) -- feeds next GEMM's B operand directly.
// MODE 1: store Out row-major f16 with bias + ReLU.
// MODE 2: store Out row-major f32 with bias (final layer).
// ---------------------------------------------------------------------------
template<int MODE>
__global__ __launch_bounds__(256)
void gcn_gemm_wmma(const _Float16* __restrict__ Aop,
                   const _Float16* __restrict__ Bop,
                   size_t bStrideB,
                   _Float16* __restrict__ outH,
                   float*    __restrict__ outF,
                   const float* __restrict__ bias) {
    constexpr int N = GCN_N;
    __shared__ _Float16 Al[2][TILE * LDSS];
    __shared__ _Float16 Bl[2][TILE * LDSS];

    const int b     = blockIdx.y;
    const int tM    = blockIdx.x & 3;
    const int tN    = blockIdx.x >> 2;
    const int tid   = threadIdx.x;
    const int lane  = tid & 31;
    const int wave  = tid >> 5;
    const int waveM = wave & 3;    // 4 waves along M
    const int waveN = wave >> 2;   // 2 waves along N
    const int lr    = lane & 15;
    const int kh    = lane >> 4;

    const _Float16* Ab = Aop + (size_t)b * N * N;
    const _Float16* Bb = Bop + (size_t)b * bStrideB;

    // Each thread stages 2x16B per tile per K-step (128 rows x 32 cols).
    const int srow = tid >> 2;       // 0..63 (and +64 on second chunk)
    const int skc  = tid & 3;        // which 8-f16 chunk of the 32-wide row

    auto stage = [&](int buf, int kk) {
        #pragma unroll
        for (int u = 0; u < 2; ++u) {
            const int row = srow + u * 64;
            const size_t ga = (size_t)(tM * TILE + row) * N + kk + skc * 8;
            const size_t gb = (size_t)(tN * TILE + row) * N + kk + skc * 8;
            const int    lo = row * LDSS + skc * 8;
#if GCN_ASYNC
            gcn_async_cp16(&Al[buf][lo], &Ab[ga]);
            gcn_async_cp16(&Bl[buf][lo], &Bb[gb]);
#else
            *(v8h*)&Al[buf][lo] = *(const v8h*)&Ab[ga];
            *(v8h*)&Bl[buf][lo] = *(const v8h*)&Bb[gb];
            // speculative prefetch of the next K tile; dropped if stale
            __builtin_prefetch(&Ab[ga + 32], 0, 1);
            __builtin_prefetch(&Bb[gb + 32], 0, 1);
#endif
        }
    };

    v8f acc[2][4];
    #pragma unroll
    for (int i = 0; i < 2; ++i)
        #pragma unroll
        for (int j = 0; j < 4; ++j)
            acc[i][j] = (v8f){0.f, 0.f, 0.f, 0.f, 0.f, 0.f, 0.f, 0.f};

    stage(0, 0);
#if GCN_ASYNC
    gcn_wait_async();
#endif
    __syncthreads();

    for (int k0 = 0; k0 < N; k0 += 32) {
        const int cur = (k0 >> 5) & 1;
        if (k0 + 32 < N) stage(cur ^ 1, k0 + 32);   // overlap next tile with compute

        // A fragment per ISA 16x32 f16 layout: lanes<16 hold K 0-7 & 16-23,
        // lanes>=16 hold K 8-15 & 24-31, row M = lane%16.
        v16h afrag[2], bfrag[4];
        #pragma unroll
        for (int i = 0; i < 2; ++i) {
            const _Float16* p = &Al[cur][(waveM * 32 + i * 16 + lr) * LDSS + kh * 8];
            v8h lo = *(const v8h*)p;
            v8h hi = *(const v8h*)(p + 16);
            #pragma unroll
            for (int e = 0; e < 8; ++e) { afrag[i][e] = lo[e]; afrag[i][8 + e] = hi[e]; }
        }
        // B fragment: lane = column n (mod 16), lanes<16 hold K 0-7 & 16-23, lanes>=16 K 8-15 & 24-31,
        // from the Bt row.
        #pragma unroll
        for (int j = 0; j < 4; ++j) {
            const _Float16* p = &Bl[cur][(waveN * 64 + j * 16 + lr) * LDSS + kh * 8];
            v8h lo = *(const v8h*)p;
            v8h hi = *(const v8h*)(p + 16);
            #pragma unroll
            for (int e = 0; e < 8; ++e) { bfrag[j][e] = lo[e]; bfrag[j][8 + e] = hi[e]; }
        }

        #pragma unroll
        for (int i = 0; i < 2; ++i)
            #pragma unroll
            for (int j = 0; j < 4; ++j)
                acc[i][j] = __builtin_amdgcn_wmma_f32_16x16x32_f16(
                    false, afrag[i], false, bfrag[j],
                    (short)0, acc[i][j], false, false);

#if GCN_ASYNC
        gcn_wait_async();    // next buffer fully landed in LDS (ASYNCcnt==0)
#endif
        __syncthreads();
    }

    // Epilogue. C/D layout: VGPR e, lane l -> M = e + 8*(l>>4), N = l&15.
    #pragma unroll
    for (int i = 0; i < 2; ++i)
        #pragma unroll
        for (int j = 0; j < 4; ++j) {
            const int m0 = tM * TILE + waveM * 32 + i * 16 + kh * 8;
            const int n  = tN * TILE + waveN * 64 + j * 16 + lr;
            if (MODE == 0) {
                // transposed store: lane's 8 accum values are contiguous in OutT row n
                v8h o;
                #pragma unroll
                for (int e = 0; e < 8; ++e) o[e] = (_Float16)acc[i][j][e];
                *(v8h*)&outH[((size_t)b * N + n) * N + m0] = o;
            } else if (MODE == 1) {
                const float bn = bias[n];
                #pragma unroll
                for (int e = 0; e < 8; ++e) {
                    float v = acc[i][j][e] + bn;
                    v = v > 0.f ? v : 0.f;
                    outH[((size_t)b * N + m0 + e) * N + n] = (_Float16)v;
                }
            } else {
                const float bn = bias[n];
                #pragma unroll
                for (int e = 0; e < 8; ++e)
                    outF[((size_t)b * N + m0 + e) * N + n] = acc[i][j][e] + bn;
            }
        }
}

// ---------------------------------------------------------------------------
extern "C" void kernel_launch(void* const* d_in, const int* in_sizes, int n_in,
                              void* d_out, int out_size, void* d_ws, size_t ws_size,
                              hipStream_t stream) {
    const float* X0  = (const float*)d_in[0];   // [64,512,512]
    const float* adj = (const float*)d_in[1];   // [64,512,512]
    const float* W1  = (const float*)d_in[2];
    const float* b1  = (const float*)d_in[3];
    const float* W2  = (const float*)d_in[4];
    const float* b2  = (const float*)d_in[5];
    const float* W3  = (const float*)d_in[6];
    const float* b3  = (const float*)d_in[7];
    float* out = (float*)d_out;

    const int N = GCN_N, B = GCN_B;
    const size_t NN  = (size_t)N * N;    // 262144
    const size_t BNN = (size_t)B * NN;   // 16M elems

    char* ws = (char*)d_ws;
    float*    r   = (float*)ws;     ws += (((size_t)B * N * sizeof(float)) + 255) & ~(size_t)255;
    _Float16* Ah  = (_Float16*)ws;  ws += BNN * 2;   // 32 MB
    _Float16* Xh  = (_Float16*)ws;  ws += BNN * 2;   // 32 MB
    _Float16* St  = (_Float16*)ws;  ws += BNN * 2;   // 32 MB
    _Float16* W1t = (_Float16*)ws;  ws += NN * 2;
    _Float16* W2t = (_Float16*)ws;  ws += NN * 2;
    _Float16* W3t = (_Float16*)ws;

    const int nb2 = (N / 32) * (N / 32);   // 256 transpose tiles

    gcn_colsum_rsqrt<<<B, 256, 0, stream>>>(adj, r);
    gcn_build_ahat<<<dim3(nb2, B), 256, 0, stream>>>(adj, r, Ah);
    gcn_cast_h4<<<(unsigned)((BNN / 4 + 255) / 256), 256, 0, stream>>>(X0, Xh, BNN / 4);
    gcn_transpose_w<<<nb2, 256, 0, stream>>>(W1, W1t);
    gcn_transpose_w<<<nb2, 256, 0, stream>>>(W2, W2t);
    gcn_transpose_w<<<nb2, 256, 0, stream>>>(W3, W3t);

    dim3 gg((N / TILE) * (N / TILE), B);   // 16 x 64 blocks

    // layer 1: St = (X @ W1)^T ; Xh = relu(A @ S + b1)
    gcn_gemm_wmma<0><<<gg, 256, 0, stream>>>(Xh, W1t, 0,  St, nullptr, nullptr);
    gcn_gemm_wmma<1><<<gg, 256, 0, stream>>>(Ah, St, NN,  Xh, nullptr, b1);
    // layer 2
    gcn_gemm_wmma<0><<<gg, 256, 0, stream>>>(Xh, W2t, 0,  St, nullptr, nullptr);
    gcn_gemm_wmma<1><<<gg, 256, 0, stream>>>(Ah, St, NN,  Xh, nullptr, b2);
    // layer 3: out = A @ S + b3 (f32, no relu)
    gcn_gemm_wmma<0><<<gg, 256, 0, stream>>>(Xh, W3t, 0,  St, nullptr, nullptr);
    gcn_gemm_wmma<2><<<gg, 256, 0, stream>>>(Ah, St, NN,  nullptr, out, b3);
}
